// PhotonicSNN_44727789420605
// MI455X (gfx1250) — compile-verified, matrix operand order checked by reading, not run on hardware
//
#include <hip/hip_runtime.h>

// ---------------------------------------------------------------------------
// Photonic SNN on MI455X (gfx1250, wave32, WMMA + async-LDS staging).
//
// Layer-2 spikes never feed back into layer 1, so the 1000-step scan factors
// into: GEMM1 -> per-neuron LIF scan -> GEMM2 -> per-neuron LIF scan.
// GEMMs run on v_wmma_f32_16x16x32_f16 with block tiles double-buffered into
// LDS via global_load_async_to_lds_b128 (ASYNCcnt) for ~64 FLOP/byte of L2.
// All 12 ds_load_b128 per K-step are issued before the 8-WMMA burst so the
// compiler can overlap LDS latency with the XDL pipeline (partial dscnt waits).
// ---------------------------------------------------------------------------

typedef __attribute__((ext_vector_type(16))) _Float16 v16h;
typedef __attribute__((ext_vector_type(8)))  _Float16 v8h;
typedef __attribute__((ext_vector_type(8)))  float    v8f;

#define N_STEPS 1000
#define N_IN    2048
#define N_HID   8192
#define M_PAD   1024   // N_STEPS padded up to the 128-row GEMM M tile

// LIF constants (DT=1e-6, TAU_MEM=0.02, TAU_REF=0.002, THRESH=1e-6)
#define LIF_DECAY 0.99995000124997917f   // exp(-DT/TAU_MEM)
#define LIF_GAIN  5.0e-5f                // DT/TAU_MEM
#define LIF_THR   1.0e-6f
#define LIF_REF   0.002f
#define LIF_DT    1.0e-6f

// --- CDNA5 async copy global->LDS (tracked by ASYNCcnt) --------------------
// LDS flat addresses truncate to the LDS byte offset (ISA 10.2 aperture rule),
// so the low 32 bits of a __shared__ pointer are the VDST operand.
__device__ __forceinline__ void async_copy_b128(const _Float16* g, _Float16* l) {
  unsigned           loff = (unsigned)(unsigned long long)l;
  unsigned long long ga   = (unsigned long long)g;
  asm volatile("global_load_async_to_lds_b128 %0, %1, off"
               :: "v"(loff), "v"(ga) : "memory");
}
__device__ __forceinline__ void wait_asynccnt_le4() {
  asm volatile("s_wait_asynccnt 0x4" ::: "memory");
}

// K index inside a 16-bit 16x32 A fragment (mirrored 32x16 B fragment)
// for (lane, half) per the CDNA5 ISA VGPR layout tables.
__device__ __forceinline__ int kof(int lane, int h) {
  return ((h >> 3) << 4) + ((lane >> 4) << 3) + (h & 7);
}

// --- input spikes fp32 [1000x2048] -> fp16 row-major padded [1024x2048] ----
__global__ void snn_conv_x(const float* __restrict__ X, _Float16* __restrict__ XH) {
  int tid = blockIdx.x * blockDim.x + threadIdx.x;   // [0, M_PAD*N_IN)
  int t = tid >> 11;                                  // row (N_IN == 2048)
  float v = (t < N_STEPS) ? X[tid] : 0.f;
  XH[tid] = (_Float16)v;
}

// --- weights fp32 row-major [K x N] -> fp16 WMMA-B-fragment-swizzled -------
// Fragment (kt, nt) covers B[kt*32 .. +31, nt*16 .. +15]; stored as
// 32 lanes x 16 halves contiguous (1 KB) so the GEMM loads one v16h/lane.
__global__ void snn_conv_w(const float* __restrict__ W, _Float16* __restrict__ WH,
                           int K, int N) {
  int tid = blockIdx.x * blockDim.x + threadIdx.x;   // K*N threads
  int within = tid & 511;
  int fragId = tid >> 9;
  int lane = within >> 4;
  int h    = within & 15;
  int nfrag = N >> 4;
  int kt = fragId / nfrag;
  int nt = fragId - kt * nfrag;
  int k = (kt << 5) + kof(lane, h);
  int n = (nt << 4) + (lane & 15);
  WH[tid] = (_Float16)W[(size_t)k * N + n];
}

// --- f16 WMMA GEMM with async-LDS double buffering -------------------------
// C[1024 x N] = A[1024 x K] (row-major f16) * Bsw (fragment-swizzled f16).
// Block = 256 threads = 8 waves as 4(M) x 2(N); block tile 128M x 128N;
// wave tile 32M x 64N (8 accumulators). Per K-step(32) the block stages
// A-tile (8 KB) + 8 B-fragments (8 KB) into LDS with 4 async b128/thread.
template <int K, int N>
__global__ __launch_bounds__(256) void snn_gemm_wmma(
    const _Float16* __restrict__ A,
    const _Float16* __restrict__ Bsw,
    float* __restrict__ C)
{
  constexpr int NFRAG = N >> 4;
  constexpr int HALF_BUF = 128 * 32 + 8 * 512;       // 8192 halves = 16 KB
  __shared__ alignas(16) _Float16 smem[2][HALF_BUF]; // 32 KB double buffer

  const int tid  = threadIdx.x;
  const int lane = tid & 31;
  const int wave = tid >> 5;
  const int mw = wave >> 1;                // 0..3
  const int nw = wave & 1;                 // 0..1
  const int m_blk = blockIdx.x * 128;
  const int n_blk = blockIdx.y * 128;
  const int row = lane & 15;
  const int kg  = lane >> 4;
  const int nb  = n_blk >> 4;              // first B fragment column index

  // Issue this thread's 4 async b128 copies for K-slice k0 into buffer p.
  auto stage = [&](int k0, int p) {
    const _Float16* gB = Bsw + ((size_t)(k0 >> 5) * NFRAG + nb) * 512;
#pragma unroll
    for (int s = 0; s < 2; ++s) {
      int c = tid + s * 256;               // 0..511 sixteen-byte chunks
      int arow = c >> 2, aseg = c & 3;
      async_copy_b128(A + (size_t)(m_blk + arow) * K + k0 + aseg * 8,
                      &smem[p][arow * 32 + aseg * 8]);
      async_copy_b128(gB + c * 8, &smem[p][4096 + c * 8]);
    }
  };

  v8f acc[2][4];
#pragma unroll
  for (int i = 0; i < 2; ++i)
#pragma unroll
    for (int j = 0; j < 4; ++j)
      acc[i][j] = (v8f){0.f, 0.f, 0.f, 0.f, 0.f, 0.f, 0.f, 0.f};

  stage(0, 0);
  int p = 0;
  for (int k0 = 0; k0 < K; k0 += 32) {
    const int kn = (k0 + 32 < K) ? k0 + 32 : 0;      // dead prefetch on last
    stage(kn, p ^ 1);
    wait_asynccnt_le4();                             // current buffer landed
    __syncthreads();

    const _Float16* bufA = &smem[p][0];
    const _Float16* bufB = &smem[p][4096];

    // Issue ALL LDS reads for this K-step first (12x ds_load_b128), ...
    v16h a[2];
#pragma unroll
    for (int i = 0; i < 2; ++i) {
      // LDS A row: 32 halves; [0:8)=K0-7 [8:16)=K8-15 [16:24)=K16-23 [24:32)=K24-31
      const _Float16* pA = bufA + (mw * 32 + i * 16 + row) * 32 + kg * 8;
      v8h lo = *(const v8h*)pA;          // K = kg*8 .. +7
      v8h hi = *(const v8h*)(pA + 16);   // K = 16 + kg*8 .. +7
      a[i] = __builtin_shufflevector(lo, hi,
                                     0, 1, 2, 3, 4, 5, 6, 7,
                                     8, 9, 10, 11, 12, 13, 14, 15);
    }
    v16h b[4];
#pragma unroll
    for (int j = 0; j < 4; ++j)
      b[j] = ((const v16h*)(bufB + (nw * 4 + j) * 512))[lane];

    // ... then fire the 8-WMMA burst (partial dscnt waits, back-to-back XDL).
#pragma unroll
    for (int j = 0; j < 4; ++j)
#pragma unroll
      for (int i = 0; i < 2; ++i)
        acc[i][j] = __builtin_amdgcn_wmma_f32_16x16x32_f16(
            false, a[i], false, b[j], (short)0, acc[i][j], false, false);

    __syncthreads();                     // done reading buf p before reuse
    p ^= 1;
  }

  // C/D layout: VGPR r -> (M = kg*8 + r, N = row) inside each 16x16 tile.
  // N,K constexpr => row/col offsets fold into immediate store offsets.
  const int m0 = m_blk + mw * 32;
  const int n0 = n_blk + nw * 64;
#pragma unroll
  for (int i = 0; i < 2; ++i) {
    float* cb = C + (size_t)(m0 + i * 16 + kg * 8) * N + n0 + row;
#pragma unroll
    for (int r = 0; r < 8; ++r)
#pragma unroll
      for (int j = 0; j < 4; ++j)
        cb[r * N + j * 16] = acc[i][j][r];
  }
}

// --- LIF scan over time, one thread per hidden neuron; emits f16 spikes ----
__global__ void snn_scan_hidden(const float* __restrict__ C1,
                                _Float16* __restrict__ S1H) {
  int j = blockIdx.x * blockDim.x + threadIdx.x;   // < N_HID
  float mem = 0.f, ref = 0.f;
  for (int t = 0; t < N_STEPS; ++t) {
    float cur = C1[(size_t)t * N_HID + j];
    mem = mem * LIF_DECAY + cur * LIF_GAIN;
    bool fire = (mem > LIF_THR) && (ref <= 0.f);
    S1H[(size_t)t * N_HID + j] = (_Float16)(fire ? 1.f : 0.f);
    mem = fire ? 0.f : mem;
    ref = fire ? LIF_REF : fmaxf(0.f, ref - LIF_DT);
  }
  for (int t = N_STEPS; t < M_PAD; ++t)            // zero GEMM2 pad rows
    S1H[(size_t)t * N_HID + j] = (_Float16)0.f;
}

// --- LIF scan for output layer; writes f32 spikes to d_out ------------------
__global__ void snn_scan_out(const float* __restrict__ C2,
                             float* __restrict__ out) {
  int i = blockIdx.x * blockDim.x + threadIdx.x;   // < N_IN
  float mem = 0.f, ref = 0.f;
  for (int t = 0; t < N_STEPS; ++t) {
    float cur = C2[(size_t)t * N_IN + i];
    mem = mem * LIF_DECAY + cur * LIF_GAIN;
    bool fire = (mem > LIF_THR) && (ref <= 0.f);
    out[(size_t)t * N_IN + i] = fire ? 1.f : 0.f;
    mem = fire ? 0.f : mem;
    ref = fire ? LIF_REF : fmaxf(0.f, ref - LIF_DT);
  }
}

extern "C" void kernel_launch(void* const* d_in, const int* in_sizes, int n_in_args,
                              void* d_out, int out_size, void* d_ws, size_t ws_size,
                              hipStream_t stream) {
  (void)in_sizes; (void)n_in_args; (void)out_size; (void)ws_size;

  const float* X  = (const float*)d_in[0];   // [1000 x 2048]
  const float* W1 = (const float*)d_in[1];   // [2048 x 8192]
  const float* W2 = (const float*)d_in[2];   // [8192 x 2048]
  float* out = (float*)d_out;                // [1000 x 2048]

  // Workspace carve-out (~124 MB total)
  char* ws = (char*)d_ws;
  _Float16* XH  = (_Float16*)ws; ws += (size_t)M_PAD * N_IN  * 2;  //   4 MB
  _Float16* W1H = (_Float16*)ws; ws += (size_t)N_IN  * N_HID * 2;  //  32 MB
  _Float16* W2H = (_Float16*)ws; ws += (size_t)N_HID * N_IN  * 2;  //  32 MB
  _Float16* S1H = (_Float16*)ws; ws += (size_t)M_PAD * N_HID * 2;  //  16 MB
  float*    C1  = (float*)ws;    ws += (size_t)M_PAD * N_HID * 4;  //  32 MB
  float*    C2  = (float*)ws;    ws += (size_t)M_PAD * N_IN  * 4;  //   8 MB

  snn_conv_x<<<(M_PAD * N_IN) / 256, 256, 0, stream>>>(X, XH);
  snn_conv_w<<<(N_IN * N_HID) / 256, 256, 0, stream>>>(W1, W1H, N_IN, N_HID);
  snn_conv_w<<<(N_HID * N_IN) / 256, 256, 0, stream>>>(W2, W2H, N_HID, N_IN);

  // GEMM1: C1[1024 x 8192] = XH * W1H
  snn_gemm_wmma<N_IN, N_HID>
      <<<dim3(M_PAD / 128, N_HID / 128), 256, 0, stream>>>(XH, W1H, C1);

  snn_scan_hidden<<<N_HID / 256, 256, 0, stream>>>(C1, S1H);

  // GEMM2: C2[1024 x 2048] = S1H * W2H
  snn_gemm_wmma<N_HID, N_IN>
      <<<dim3(M_PAD / 128, N_IN / 128), 256, 0, stream>>>(S1H, W2H, C2);

  snn_scan_out<<<N_IN / 256, 256, 0, stream>>>(C2, out);
}
